// SPCS_9363028705863
// MI455X (gfx1250) — matrix operand order repaired and checked by reading, not compiled, not run on hardware
//
#include <hip/hip_runtime.h>
#include <math.h>

// ---------------------------------------------------------------------------
// Problem constants (from the reference)
// ---------------------------------------------------------------------------
#define BB     8
#define DIMC   256
#define D3     768
#define HH     128
#define WW     128
#define NPIX   16384          // H*W
#define NHEADS 4
#define CPH    64             // channels per head

// ---------------------------------------------------------------------------
// Workspace layout (float offsets). Total ~201.6M floats (~807 MB).
// ---------------------------------------------------------------------------
static constexpr size_t OFF_QKV  = 0;                                // [B][768][N] f32
static constexpr size_t SZ_QKV   = (size_t)BB * D3 * NPIX;           // 100,663,296
static constexpr size_t OFF_QKVD = OFF_QKV + SZ_QKV;                 // [B][768][N] f32
static constexpr size_t OFF_S    = OFF_QKVD + SZ_QKV;                // [B][4][64][64] f32
static constexpr size_t SZ_S     = (size_t)BB * NHEADS * CPH * CPH;  // 131072
static constexpr size_t OFF_NORM = OFF_S + SZ_S;                     // [B][512] sum(v^2)
static constexpr size_t SZ_NORM  = (size_t)BB * 512;                 // 4096
static constexpr size_t OFF_GSUM = OFF_NORM + SZ_NORM;               // 1 float (+pad)
static constexpr size_t OFF_WHQ  = OFF_GSUM + 16;                    // f16 [768][256]
static constexpr size_t SZ_WHQ_F = (size_t)D3 * DIMC / 2;            // halves->floats
static constexpr size_t OFF_WHG  = OFF_WHQ + SZ_WHQ_F;               // f16 [128][256]

// ---------------------------------------------------------------------------
// WMMA vector types (wave32, gfx1250)
// ---------------------------------------------------------------------------
typedef _Float16 v16h __attribute__((ext_vector_type(16)));
typedef _Float16 v8h  __attribute__((ext_vector_type(8)));
typedef float    v8f  __attribute__((ext_vector_type(8)));
typedef float    v2f  __attribute__((ext_vector_type(2)));

#define CAT16(lo, hi) __builtin_shufflevector((lo), (hi), 0,1,2,3,4,5,6,7,8,9,10,11,12,13,14,15)

// ===========================================================================
// k0: init scratch + convert weights to f16
// ===========================================================================
__global__ __launch_bounds__(256) void k0_init(const float* __restrict__ wqkv,
                                               const float* __restrict__ gw1,
                                               _Float16* __restrict__ whq,
                                               _Float16* __restrict__ whg,
                                               float* __restrict__ S,
                                               float* __restrict__ normsq,
                                               float* __restrict__ gsum) {
  int i = blockIdx.x * 256 + threadIdx.x;
  if (i < D3 * DIMC)   whq[i] = (_Float16)wqkv[i];
  if (i < 128 * DIMC)  whg[i] = (_Float16)gw1[i];
  if (i < (int)SZ_S)   S[i] = 0.0f;
  if (i < (int)SZ_NORM) normsq[i] = 0.0f;
  if (i == 0)          gsum[0] = 0.0f;
}

// ===========================================================================
// k1: fused conv1x1 qkv GEMM (768 rows) + gate GEMM (128 rows) per 64-pixel
//     tile.  f16 WMMA 16x16x32, f32 accumulate.  2048 blocks x 256 threads.
// ===========================================================================
#define XT_STRIDE 264   // 256 + 8 halves pad
#define G1_STRIDE 72    // 64 + 8 halves pad

__global__ __launch_bounds__(256) void k1_qkv_gate(const float* __restrict__ x,
                                                   const _Float16* __restrict__ whq,
                                                   const _Float16* __restrict__ whg,
                                                   const float* __restrict__ gb1,
                                                   const float* __restrict__ gw2,
                                                   const float* __restrict__ gb2,
                                                   float* __restrict__ qkv,
                                                   float* __restrict__ gsum) {
  __shared__ __align__(16) _Float16 xt[64 * XT_STRIDE];   // [pixel][channel] f16
  __shared__ __align__(16) _Float16 g1s[128 * G1_STRIDE]; // relu(W1@x) f16
  __shared__ float red[64];

  const int tid  = threadIdx.x;
  const int pix0 = blockIdx.x * 64;
  const int b    = pix0 >> 14;          // / NPIX
  const int p0   = pix0 & (NPIX - 1);

  // ---- stage x tile (f32 global -> f16 LDS) ----
  for (int i = tid; i < 64 * DIMC; i += 256) {
    int pp = i & 63;
    int c  = i >> 6;
    float v = x[((size_t)b * DIMC + c) * NPIX + p0 + pp];
    xt[pp * XT_STRIDE + c] = (_Float16)v;
  }
  __syncthreads();

  const int lane = tid & 31;
  const int wid  = tid >> 5;            // 8 waves
  const int l16  = lane & 15;
  const int lh   = lane >> 4;           // 0: lanes 0-15, 1: lanes 16-31

  // ---- qkv GEMM: 48 o-tiles x 4 p-tiles, 8 k-steps of 32 ----
  for (int j = 0; j < 6; ++j) {
    const int ot   = wid + 8 * j;
    const int arow = ot * 16 + l16;
    for (int pt = 0; pt < 4; ++pt) {
      v8f acc = {};
      const int col = pt * 16 + l16;
#pragma unroll
      for (int ks = 0; ks < 8; ++ks) {
        const int k0 = ks * 32;
        // A frag: lane-half K layout {0-7,16-23} / {8-15,24-31}
        const int ka = k0 + lh * 8;
        v8h alo = *(const v8h*)(whq + arow * DIMC + ka);
        v8h ahi = *(const v8h*)(whq + arow * DIMC + ka + 16);
        v16h a = CAT16(alo, ahi);
        // B frag: lane-half K layout {0-7,16-23} / {8-15,24-31}, N = column = pixel
        const int kb = k0 + lh * 8;
        v8h blo = *(const v8h*)(xt + col * XT_STRIDE + kb);
        v8h bhi = *(const v8h*)(xt + col * XT_STRIDE + kb + 16);
        v16h bf = CAT16(blo, bhi);
        acc = __builtin_amdgcn_wmma_f32_16x16x32_f16(false, a, false, bf,
                                                     (short)0, acc, false, false);
      }
      const int pgl = p0 + pt * 16 + l16;
#pragma unroll
      for (int r = 0; r < 8; ++r) {
        int o = ot * 16 + r + lh * 8;
        qkv[((size_t)b * D3 + o) * NPIX + pgl] = acc[r];
      }
    }
  }

  // ---- gate GEMM: 8 o-tiles, one per wave ----
  {
    const int got  = wid;
    const int arow = got * 16 + l16;
    for (int pt = 0; pt < 4; ++pt) {
      v8f acc = {};
      const int col = pt * 16 + l16;
#pragma unroll
      for (int ks = 0; ks < 8; ++ks) {
        const int k0 = ks * 32;
        const int ka = k0 + lh * 8;
        v8h alo = *(const v8h*)(whg + arow * DIMC + ka);
        v8h ahi = *(const v8h*)(whg + arow * DIMC + ka + 16);
        v16h a = CAT16(alo, ahi);
        const int kb = k0 + lh * 8;
        v8h blo = *(const v8h*)(xt + col * XT_STRIDE + kb);
        v8h bhi = *(const v8h*)(xt + col * XT_STRIDE + kb + 16);
        v16h bf = CAT16(blo, bhi);
        acc = __builtin_amdgcn_wmma_f32_16x16x32_f16(false, a, false, bf,
                                                     (short)0, acc, false, false);
      }
#pragma unroll
      for (int r = 0; r < 8; ++r) {
        int o = got * 16 + r + lh * 8;
        float v = acc[r] + gb1[o];
        g1s[o * G1_STRIDE + col] = (_Float16)fmaxf(v, 0.0f);
      }
    }
  }
  __syncthreads();

  // ---- gate second layer + sigmoid + block partial mean ----
  if (tid < 64) {
    float s = gb2[0];
    for (int m = 0; m < 128; ++m)
      s += gw2[m] * (float)g1s[m * G1_STRIDE + tid];
    red[tid] = 1.0f / (1.0f + expf(-s));
  }
  __syncthreads();
  if (tid == 0) {
    float t = 0.0f;
    for (int i = 0; i < 64; ++i) t += red[i];
    atomicAdd(gsum, t);
  }
}

// ===========================================================================
// k2: 3x3 depthwise conv (SAME, zero pad) + sum-of-squares for q,k channels.
//     Grid = B*768*(H/8), 256 threads; each block = one (b,o) x 8 rows.
// ===========================================================================
__global__ __launch_bounds__(256) void k2_dwconv(const float* __restrict__ qkv,
                                                 const float* __restrict__ wdw,
                                                 float* __restrict__ qkvd,
                                                 float* __restrict__ normsq) {
  __shared__ float sred[256];
  const int tid = threadIdx.x;
  const int yt  = blockIdx.x & 15;
  const int bo  = blockIdx.x >> 4;
  const int o   = bo % D3;
  const int b   = bo / D3;

  float wv[9];
#pragma unroll
  for (int i = 0; i < 9; ++i) wv[i] = wdw[o * 9 + i];

  const float* src = qkv  + ((size_t)b * D3 + o) * NPIX;
  float*       dst = qkvd + ((size_t)b * D3 + o) * NPIX;

  const int y  = yt * 8 + (tid >> 5);
  const int x0 = (tid & 31) * 4;

  float ssq = 0.0f;
#pragma unroll
  for (int xi = 0; xi < 4; ++xi) {
    const int xx = x0 + xi;
    float acc = 0.0f;
#pragma unroll
    for (int dy = -1; dy <= 1; ++dy) {
      const int yy = y + dy;
      if (yy < 0 || yy >= HH) continue;
#pragma unroll
      for (int dx = -1; dx <= 1; ++dx) {
        const int x2 = xx + dx;
        if (x2 < 0 || x2 >= WW) continue;
        acc += src[yy * WW + x2] * wv[(dy + 1) * 3 + (dx + 1)];
      }
    }
    dst[y * WW + xx] = acc;
    ssq += acc * acc;
  }

  // block reduction of sum-of-squares; only q (o<256) / k (256<=o<512) channels
  sred[tid] = ssq;
  __syncthreads();
  for (int s = 128; s > 0; s >>= 1) {
    if (tid < s) sred[tid] += sred[tid + s];
    __syncthreads();
  }
  if (tid == 0 && o < 512) atomicAdd(&normsq[b * 512 + o], sred[0]);
}

// ===========================================================================
// k4: Gram matrix S[b,h] = q @ k^T over n=16384, f32 WMMA 16x16x4.
//     Grid = B*HEADS*16 K-chunks (1024 each); 256 threads; atomic partials.
// ===========================================================================
#define QT_STRIDE 68   // 64 + 4 floats pad

__global__ __launch_bounds__(256) void k4_gram(const float* __restrict__ qkvd,
                                               float* __restrict__ S) {
  __shared__ __align__(16) float qt[64 * QT_STRIDE];
  __shared__ __align__(16) float kt[64 * QT_STRIDE];

  const int tid   = threadIdx.x;
  const int chunk = blockIdx.x & 15;
  const int bh    = blockIdx.x >> 4;
  const int h     = bh & 3;
  const int b     = bh >> 2;

  const int lane = tid & 31;
  const int wid  = tid >> 5;
  const int l16  = lane & 15;
  const int lh   = lane >> 4;

  const int ct  = wid & 3;          // c-tile (rows of q)
  const int dt0 = (wid >> 2) * 2;   // first of two d-tiles (rows of k)

  v8f acc0 = {};
  v8f acc1 = {};

  const size_t qbase = ((size_t)b * D3 + h * CPH) * NPIX;
  const size_t kbase = ((size_t)b * D3 + 256 + h * CPH) * NPIX;

  for (int kb = 0; kb < 16; ++kb) {
    const int nbase = chunk * 1024 + kb * 64;
    // cooperative coalesced load of 64x64 q and k tiles
    for (int i = tid; i < 64 * 64; i += 256) {
      int row = i >> 6, col = i & 63;
      qt[row * QT_STRIDE + col] = qkvd[qbase + (size_t)row * NPIX + nbase + col];
      kt[row * QT_STRIDE + col] = qkvd[kbase + (size_t)row * NPIX + nbase + col];
    }
    __syncthreads();

#pragma unroll
    for (int kk = 0; kk < 64; kk += 4) {
      // A frag (q): lane-half K {0,1}/{2,3}
      v2f a;
      a.x = qt[(ct * 16 + l16) * QT_STRIDE + kk + lh * 2];
      a.y = qt[(ct * 16 + l16) * QT_STRIDE + kk + lh * 2 + 1];
      // B frags (k^T): element (K,N) = kt[N][K]
      v2f b0, b1;
      b0.x = kt[(dt0 * 16 + l16) * QT_STRIDE + kk + lh * 2];
      b0.y = kt[(dt0 * 16 + l16) * QT_STRIDE + kk + lh * 2 + 1];
      b1.x = kt[((dt0 + 1) * 16 + l16) * QT_STRIDE + kk + lh * 2];
      b1.y = kt[((dt0 + 1) * 16 + l16) * QT_STRIDE + kk + lh * 2 + 1];
      acc0 = __builtin_amdgcn_wmma_f32_16x16x4_f32(false, a, false, b0,
                                                   (short)0, acc0, false, false);
      acc1 = __builtin_amdgcn_wmma_f32_16x16x4_f32(false, a, false, b1,
                                                   (short)0, acc1, false, false);
    }
    __syncthreads();
  }

  float* Sbh = S + (size_t)bh * CPH * CPH;
#pragma unroll
  for (int r = 0; r < 8; ++r) {
    int c = ct * 16 + r + lh * 8;
    atomicAdd(&Sbh[c * CPH + dt0 * 16 + l16], acc0[r]);
    atomicAdd(&Sbh[c * CPH + (dt0 + 1) * 16 + l16], acc1[r]);
  }
}

// ===========================================================================
// k5: scale by temperature / (||q_c|| ||k_d||), dynamic top-k mask, softmax,
//     fold (attn1+attn2+attn3+attn4).  32 blocks x 64 threads, in place.
// ===========================================================================
__global__ __launch_bounds__(64) void k5_attnpost(float* __restrict__ S,
                                                  const float* __restrict__ normsq,
                                                  const float* __restrict__ temp,
                                                  const float* __restrict__ a1,
                                                  const float* __restrict__ a2,
                                                  const float* __restrict__ a3,
                                                  const float* __restrict__ a4,
                                                  const float* __restrict__ gsum) {
  __shared__ float nks[64];
  const int bh = blockIdx.x;
  const int h  = bh & 3;
  const int b  = bh >> 2;
  const int c  = threadIdx.x;

  // dynamic_k from the gate mean (same value in every block/thread)
  const float gmean = gsum[0] / (float)((size_t)BB * NPIX);
  int dk = (int)floorf((float)CPH * gmean);
  dk = dk < 1 ? 1 : (dk > CPH ? CPH : dk);

  const float asum = a1[0] + a2[0] + a3[0] + a4[0];
  const float tmp  = temp[h];

  nks[c] = fmaxf(sqrtf(fmaxf(normsq[b * 512 + 256 + h * CPH + c], 0.0f)), 1e-12f);
  __syncthreads();

  const float nq = fmaxf(sqrtf(fmaxf(normsq[b * 512 + h * CPH + c], 0.0f)), 1e-12f);
  float* row = S + ((size_t)bh * CPH + c) * CPH;

  float r[64];
  float m = -INFINITY;
  for (int d = 0; d < CPH; ++d) {
    float v = row[d] * tmp / (nq * nks[d]);
    r[d] = v;
    m = fmaxf(m, v);
  }

  // dk-th largest (selection; ties negligible per reference note)
  float prev = INFINITY;
  for (int t = 0; t < dk; ++t) {
    float cur = -INFINITY;
    for (int d = 0; d < CPH; ++d) {
      float v = r[d];
      if (v < prev && v > cur) cur = v;
    }
    prev = cur;
  }
  const float thr = prev;

  float s = 0.0f;
  for (int d = 0; d < CPH; ++d)
    if (r[d] >= thr) s += expf(r[d] - m);
  const float inv = asum / s;

  for (int d = 0; d < CPH; ++d)
    row[d] = (r[d] >= thr) ? expf(r[d] - m) * inv : 0.0f;
}

// ===========================================================================
// k6: out = attn @ v, f32 WMMA 16x16x4.  Grid = 32*(N/128) blocks.
// ===========================================================================
#define VT_STRIDE 132  // 128 + 4 floats pad

__global__ __launch_bounds__(256) void k6_av(const float* __restrict__ attn,
                                             const float* __restrict__ qkvd,
                                             float* __restrict__ out) {
  __shared__ __align__(16) float at[64 * QT_STRIDE];
  __shared__ __align__(16) float vt[64 * VT_STRIDE];

  const int tid   = threadIdx.x;
  const int nc    = blockIdx.x & 127;
  const int bh    = blockIdx.x >> 7;
  const int h     = bh & 3;
  const int b     = bh >> 2;
  const int nbase = nc * 128;

  // attn tile (64x64) and v tile (64x128)
  for (int i = tid; i < 64 * 64; i += 256) {
    int row = i >> 6, col = i & 63;
    at[row * QT_STRIDE + col] = attn[(size_t)bh * CPH * CPH + i];
  }
  const size_t vbase = ((size_t)b * D3 + 512 + h * CPH) * NPIX;
  for (int i = tid; i < 64 * 128; i += 256) {
    int row = i >> 7, col = i & 127;
    vt[row * VT_STRIDE + col] = qkvd[vbase + (size_t)row * NPIX + nbase + col];
  }
  __syncthreads();

  const int lane = tid & 31;
  const int wid  = tid >> 5;
  const int l16  = lane & 15;
  const int lh   = lane >> 4;

  const int mt  = wid & 3;        // c-tile
  const int nt0 = wid >> 2;       // n-tiles nt0, nt0+2, nt0+4, nt0+6

  for (int jj = 0; jj < 4; ++jj) {
    const int nt = nt0 + 2 * jj;
    v8f acc = {};
#pragma unroll
    for (int kk = 0; kk < 64; kk += 4) {
      v2f a;
      a.x = at[(mt * 16 + l16) * QT_STRIDE + kk + lh * 2];
      a.y = at[(mt * 16 + l16) * QT_STRIDE + kk + lh * 2 + 1];
      v2f bf;
      bf.x = vt[(kk + lh * 2) * VT_STRIDE + nt * 16 + l16];
      bf.y = vt[(kk + lh * 2 + 1) * VT_STRIDE + nt * 16 + l16];
      acc = __builtin_amdgcn_wmma_f32_16x16x4_f32(false, a, false, bf,
                                                  (short)0, acc, false, false);
    }
    const int pgl = nbase + nt * 16 + l16;
#pragma unroll
    for (int r = 0; r < 8; ++r) {
      int c = mt * 16 + r + lh * 8;
      out[((size_t)b * DIMC + h * CPH + c) * NPIX + pgl] = acc[r];
    }
  }
}

// ===========================================================================
// launch
// ===========================================================================
extern "C" void kernel_launch(void* const* d_in, const int* in_sizes, int n_in,
                              void* d_out, int out_size, void* d_ws, size_t ws_size,
                              hipStream_t stream) {
  const float* x     = (const float*)d_in[0];
  const float* wqkv  = (const float*)d_in[1];
  const float* wdw   = (const float*)d_in[2];
  const float* temp  = (const float*)d_in[3];
  const float* a1    = (const float*)d_in[4];
  const float* a2    = (const float*)d_in[5];
  const float* a3    = (const float*)d_in[6];
  const float* a4    = (const float*)d_in[7];
  const float* gw1   = (const float*)d_in[8];
  const float* gb1   = (const float*)d_in[9];
  const float* gw2   = (const float*)d_in[10];
  const float* gb2   = (const float*)d_in[11];
  float*       out   = (float*)d_out;

  float* ws      = (float*)d_ws;
  float* qkv     = ws + OFF_QKV;
  float* qkvd    = ws + OFF_QKVD;
  float* S       = ws + OFF_S;
  float* normsq  = ws + OFF_NORM;
  float* gsum    = ws + OFF_GSUM;
  _Float16* whq  = (_Float16*)(ws + OFF_WHQ);
  _Float16* whg  = (_Float16*)(ws + OFF_WHG);

  // 0) init + weight conversion
  k0_init<<<(D3 * DIMC + 255) / 256, 256, 0, stream>>>(wqkv, gw1, whq, whg,
                                                       S, normsq, gsum);
  // 1) fused qkv conv1x1 + gate (f16 WMMA)
  k1_qkv_gate<<<(BB * NPIX) / 64, 256, 0, stream>>>(x, whq, whg, gb1, gw2, gb2,
                                                    qkv, gsum);
  // 2) depthwise 3x3 + q/k sum-of-squares
  k2_dwconv<<<BB * D3 * (HH / 8), 256, 0, stream>>>(qkv, wdw, qkvd, normsq);
  // 3) Gram matrix q@k^T (f32 WMMA, K-chunked with atomic partials)
  k4_gram<<<BB * NHEADS * 16, 256, 0, stream>>>(qkvd, S);
  // 4) scale/top-k/softmax (+fold attn scalar sum)
  k5_attnpost<<<BB * NHEADS, 64, 0, stream>>>(S, normsq, temp, a1, a2, a3, a4, gsum);
  // 5) attn @ v -> output (f32 WMMA)
  k6_av<<<BB * NHEADS * (NPIX / 128), 256, 0, stream>>>(S, qkvd, out);
}
